// DWDBNet_39359080300629
// MI455X (gfx1250) — matrix operand order repaired and checked by reading, not compiled, not run on hardware
//
#include <hip/hip_runtime.h>
#include <cstdint>

// ---------------------------------------------------------------------------
// MI455X (gfx1250) pipeline:
//   prep_panel  : pack conv weights -> bf16 A-panels ALREADY in WMMA per-lane
//                 fragment layout [mtile][kchunk][lane][16 halves]
//   conv1_fused : direct 1->32 conv + copy x into feat ch0 (bf16)
//   wmma_conv   : implicit-GEMM conv3x3. Halo rows -> LDS, im2col panel built
//                 in LDS with shift/mask addressing (division-free), inner
//                 loop = contiguous v16bf fragment loads + v_wmma_f32_16x16x32_bf16
//   pool_mean / gen_dw : squeeze-excite dynamic-kernel generator
//   dyn_dw7     : per-sample depthwise 7x7 + bias + lrelu
//   wmma_conv (fp32-out mode) : final 128->3 conv into d_out
// Activations kept in bf16 so the 134MB feature map lives in the 192MB L2.
// ---------------------------------------------------------------------------

typedef unsigned short bfr;   // raw bf16 bits
typedef __attribute__((ext_vector_type(16))) __bf16 v16bf;
typedef __attribute__((ext_vector_type(8)))  float  v8f;

#define ROWP 72   // padded LDS row pitch (needs 66)

__device__ __forceinline__ bfr f2bf(float f) {
  union { float f; unsigned int u; } c; c.f = f;
  unsigned int u = c.u;
  u += 0x7FFFu + ((u >> 16) & 1u);           // round-to-nearest-even
  return (bfr)(u >> 16);
}
__device__ __forceinline__ float bf2f(bfr h) {
  union { unsigned int u; float f; } c; c.u = ((unsigned int)h) << 16;
  return c.f;
}
__device__ __forceinline__ float lrelu(float v) { return v >= 0.f ? v : 0.2f * v; }

// ---------------------------------------------------------------------------
// Pack OIHW fp32 weights into bf16 A-panels in WMMA fragment layout:
//   dst[((mt*nChunks + chunk)*32 + lane)*16 + h]
// Contiguous K layout: lanes<16 hold K = h, lanes>=16 hold K = 16+h (matches B).
// All the divide-by-9 address math lives here (tiny kernel), not in the hot loop.
// ---------------------------------------------------------------------------
__global__ void prep_panel(const float* __restrict__ w, int Cout, int Cin,
                           int mTiles, int nChunks, bfr* __restrict__ dst) {
  int idx = blockIdx.x * blockDim.x + threadIdx.x;
  int total = mTiles * nChunks * 512;          // 32 lanes * 16 halves
  if (idx >= total) return;
  int h = idx & 15;
  int lane = (idx >> 4) & 31;
  int blk = idx >> 9;                          // mt*nChunks + chunk
  int chunk = blk % nChunks, mt = blk / nChunks;
  bool hiL = lane >= 16;
  int m = (mt << 4) + (lane & 15);
  int k = (chunk << 5) + h + (hiL ? 16 : 0);
  float v = 0.f;
  if (m < Cout && k < Cin * 9) {
    int c = k / 9, t = k - c * 9;
    v = w[(m * Cin + c) * 9 + t];
  }
  dst[idx] = f2bf(v);
}

// ---------------------------------------------------------------------------
// conv1: 1->32 direct (K=9 too small for WMMA); also writes feat ch0 = x.
// ---------------------------------------------------------------------------
__global__ void __launch_bounds__(256)
conv1_fused(const float* __restrict__ x, const float* __restrict__ w,
            const float* __restrict__ bias, bfr* __restrict__ feat) {
  __shared__ float sw[288];
  __shared__ float sb[32];
  __shared__ float rows[3][260];
  int y = blockIdx.x, b = blockIdx.y, tid = threadIdx.x;
  for (int i = tid; i < 288; i += 256) sw[i] = w[i];
  if (tid < 32) sb[tid] = bias[tid];
  for (int i = tid; i < 3 * 258; i += 256) {
    int r = i / 258, col = i - r * 258;
    int yy = y + r - 1, xx = col - 1;
    float v = 0.f;
    if ((unsigned)yy < 256u && (unsigned)xx < 256u)
      v = x[((size_t)b << 16) + (yy << 8) + xx];
    rows[r][col] = v;
  }
  __syncthreads();
  int xc = tid;
  size_t base = ((size_t)(b * 128) << 16) + (y << 8) + xc;
  feat[base] = f2bf(rows[1][xc + 1]);               // channel 0 = input copy
#pragma unroll 4
  for (int oc = 0; oc < 32; ++oc) {
    float acc = sb[oc];
#pragma unroll
    for (int ky = 0; ky < 3; ++ky)
#pragma unroll
      for (int kx = 0; kx < 3; ++kx)
        acc += sw[oc * 9 + ky * 3 + kx] * rows[ky][xc + kx];
    feat[base + ((size_t)(1 + oc) << 16)] = f2bf(lrelu(acc));
  }
}

// ---------------------------------------------------------------------------
// WMMA implicit-GEMM 3x3 conv.  M=coutPad, K=Kp (=32*nChunks), N=64 px/row.
// LDS layout: [sA fragment panel][sB im2col panel][halo rows]
//   sB: [chunk][col 0..63][kk 0..31] halves -> a lane's B fragment
//       (kk = h + 16*hiLane, contiguous in h) is one v16bf load.
// ---------------------------------------------------------------------------
__global__ void __launch_bounds__(256)
wmma_conv(const bfr* __restrict__ featIn, int cin, int Kp,
          const bfr* __restrict__ Apanel, const float* __restrict__ bias,
          int cout, int coutPad,
          bfr* __restrict__ outBf, int chBase,
          float* __restrict__ outF, int outC) {
  extern __shared__ char smem[];
  const int nChunks = Kp >> 5;
  const int aBytes = coutPad * Kp * 2;
  bfr* sA = (bfr*)smem;
  bfr* sB = (bfr*)(smem + aBytes);
  bfr* sRows = (bfr*)(smem + aBytes + Kp * 128);   // Kp*64 halves im2col
  const int x0 = blockIdx.x * 64, y = blockIdx.y, b = blockIdx.z;
  const int tid = threadIdx.x;

  // ---- A panel -> LDS (TDM; cooperative copy fallback) ----------------
#if defined(__gfx1250__) && __has_builtin(__builtin_amdgcn_tensor_load_to_lds)
  if (tid < 32) {  // one wave issues the TDM descriptor
    unsigned int ndw = (unsigned int)aBytes >> 2;  // dwords
    unsigned long long ga = (unsigned long long)(uintptr_t)Apanel;
    unsigned int lds = (unsigned int)(uintptr_t)sA;
    typedef unsigned int u32x4 __attribute__((ext_vector_type(4)));
    typedef int i32x4 __attribute__((ext_vector_type(4)));
    typedef int i32x8 __attribute__((ext_vector_type(8)));
    u32x4 g0; i32x8 g1;
    i32x4 g2 = {0, 0, 0, 0}, g3 = {0, 0, 0, 0};
    i32x8 g4 = {0, 0, 0, 0, 0, 0, 0, 0};
    g0[0] = 1u;                                             // count=1 (user)
    g0[1] = lds;                                            // lds_addr
    g0[2] = (unsigned int)(ga & 0xFFFFFFFFull);             // global lo
    g0[3] = (unsigned int)((ga >> 32) & 0x01FFFFFFull) | (2u << 30); // type=2
    g1[0] = (int)(2u << 16);                                // data_size=4B
    g1[1] = (int)((ndw & 0xFFFFu) << 16);                   // tensor_dim0 lo
    g1[2] = (int)(((ndw >> 16) & 0xFFFFu) | (1u << 16));    // dim0 hi|dim1=1
    g1[3] = (int)((ndw & 0xFFFFu) << 16);                   // tile_dim0
    g1[4] = 1;                                              // tile_dim1=1
    g1[5] = (int)ndw;                                       // dim0_stride lo
    g1[6] = (int)((ndw >> 16) & 0xFFFFu);                   // dim0_stride hi
    g1[7] = 0;
    __builtin_amdgcn_tensor_load_to_lds(g0, g1, g2, g3, g4, 0);
    __builtin_amdgcn_s_wait_tensorcnt(0);
  }
#else
  {
    int totA = aBytes >> 2;
    unsigned int* d = (unsigned int*)sA;
    const unsigned int* s = (const unsigned int*)Apanel;
    for (int i = tid; i < totA; i += 256) d[i] = s[i];
  }
#endif

  // ---- stage 3 halo rows x cin channels (66 px wide), division-free ---
  {
    const int col = tid & 63, grp = tid >> 6;   // 4 channel groups
    for (int c = grp; c < cin; c += 4) {
      size_t gbase = ((size_t)(b * 128 + c) << 16);
#pragma unroll
      for (int r = 0; r < 3; ++r) {
        int yy = y + r - 1;
        bool yok = (unsigned)yy < 256u;
        int rowOff = (c * 3 + r) * ROWP;
        int xx = x0 + col - 1;
        sRows[rowOff + col] =
            (yok && (unsigned)xx < 256u) ? featIn[gbase + (yy << 8) + xx] : (bfr)0;
        if (col < 2) {
          int xx2 = x0 + col + 63;              // col+64-1
          sRows[rowOff + col + 64] =
              (yok && (unsigned)xx2 < 256u) ? featIn[gbase + (yy << 8) + xx2] : (bfr)0;
        }
      }
    }
  }
  __syncthreads();

  // ---- build im2col panel sB[chunk][col][kk], shift/mask addressing ----
  {
    const int col = tid & 63, grp = tid >> 6;
    for (int c = grp; c < cin; c += 4) {
      const bfr* rp = &sRows[c * 3 * ROWP + col];
      int kbase = c * 9;
#pragma unroll
      for (int t = 0; t < 9; ++t) {             // ky,kx constant after unroll
        int k = kbase + t;
        int ky = t / 3, kx = t - ky * 3;
        sB[((k >> 5) << 11) + (col << 5) + (k & 31)] = rp[ky * ROWP + kx];
      }
    }
    int Kreal = cin * 9;
    int padTot = (Kp - Kreal) << 6;             // (Kp-Kreal)*64
    for (int i = tid; i < padTot; i += 256) {
      int k = Kreal + (i >> 6), c2 = i & 63;
      sB[((k >> 5) << 11) + (c2 << 5) + (k & 31)] = 0;
    }
  }
  __syncthreads();

  // ---- WMMA loop: pure contiguous fragment loads ----------------------
  const int wave = tid >> 5, lane = tid & 31;
  const int mTiles = coutPad >> 4;
  if (wave < mTiles * 4) {
    const int mt = wave % mTiles, nt = wave / mTiles;
    const int m0 = mt << 4;
    const bool hiL = lane >= 16;
    const int l = lane & 15;
    const int colBase = nt * 16 + l;
    const v16bf* pAv = (const v16bf*)sA;        // [blk][lane] v16bf units
    const v16bf* pBv = (const v16bf*)sB;        // [(chunk*64+col)*2 + hiL]
    int aIdx = (mt * nChunks) * 32 + lane;
    int bIdx = colBase * 2 + (hiL ? 1 : 0);
    v8f acc = {0.f, 0.f, 0.f, 0.f, 0.f, 0.f, 0.f, 0.f};
    for (int chunk = 0; chunk < nChunks; ++chunk) {
      v16bf av = pAv[aIdx];
      v16bf bv = pBv[bIdx];
      acc = __builtin_amdgcn_wmma_f32_16x16x32_bf16(
          false, av, false, bv, (short)0, acc, false, false);
      aIdx += 32;
      bIdx += 128;                              // 64 cols * 2 v16bf per chunk
    }

    // C/D layout: VGPR r holds M=r (lanes<16) / M=r+8 (lanes>=16), N=lane&15
    const int n = x0 + colBase;
#pragma unroll
    for (int r = 0; r < 8; ++r) {
      int m = m0 + r + (hiL ? 8 : 0);
      if (m < cout) {
        float v = acc[r] + bias[m];
        if (outF) {
          outF[((size_t)(b * outC + m) << 16) + (y << 8) + n] = v;
        } else {
          outBf[((size_t)(b * 128 + chBase + m) << 16) + (y << 8) + n] =
              f2bf(lrelu(v));
        }
      }
    }
  }
}

// ---------------------------------------------------------------------------
// Global average pool: one block per (b,c) plane.
// ---------------------------------------------------------------------------
__global__ void __launch_bounds__(256)
pool_mean(const bfr* __restrict__ feat, float* __restrict__ pooled) {
  __shared__ float red[256];
  int bc = blockIdx.x, tid = threadIdx.x;
  const bfr* p = feat + ((size_t)bc << 16);
  float s = 0.f;
  for (int i = tid; i < 65536; i += 256) s += bf2f(p[i]);
  red[tid] = s;
  __syncthreads();
  for (int off = 128; off > 0; off >>= 1) {
    if (tid < off) red[tid] += red[tid + off];
    __syncthreads();
  }
  if (tid == 0) pooled[bc] = red[0] * (1.f / 65536.f);
}

// ---------------------------------------------------------------------------
// Dynamic kernel generator: pooled[8,128] -> relu(BN(fc1)) -> fc2 -> wdw[8,6272]
// ---------------------------------------------------------------------------
__global__ void __launch_bounds__(256)
gen_dw(const float* __restrict__ pooled, const float* __restrict__ w1,
       const float* __restrict__ g, const float* __restrict__ be,
       const float* __restrict__ w2, const float* __restrict__ b2,
       float* __restrict__ wdw) {
  __shared__ float sp[1024];
  __shared__ float sh[256];
  int tid = threadIdx.x;
  for (int i = tid; i < 1024; i += 256) sp[i] = pooled[i];
  __syncthreads();
  {
    int b = tid >> 5, j = tid & 31;
    float s = 0.f;
    for (int k = 0; k < 128; ++k) s += sp[b * 128 + k] * w1[j * 128 + k];
    float h = s * g[j] + be[j];
    sh[b * 32 + j] = h > 0.f ? h : 0.f;
  }
  __syncthreads();
  for (int i = tid; i < 8 * 6272; i += 256) {
    int b = i / 6272, o = i - b * 6272;
    float s = b2[o];
#pragma unroll 8
    for (int j = 0; j < 32; ++j) s += sh[b * 32 + j] * w2[o * 32 + j];
    wdw[i] = s;
  }
}

// ---------------------------------------------------------------------------
// Per-sample depthwise 7x7 + bias + lrelu. One block per (y,c,b) row.
// ---------------------------------------------------------------------------
__global__ void __launch_bounds__(256)
dyn_dw7(const bfr* __restrict__ feat, const float* __restrict__ wdw,
        const float* __restrict__ dwb, bfr* __restrict__ out) {
  __shared__ float sw[49];
  __shared__ bfr rows[7][264];
  int y = blockIdx.x, c = blockIdx.y, b = blockIdx.z;
  int tid = threadIdx.x;
  if (tid < 49) sw[tid] = wdw[(size_t)(b * 128 + c) * 49 + tid];
  const bfr* plane = feat + ((size_t)(b * 128 + c) << 16);
  for (int i = tid; i < 7 * 262; i += 256) {
    int r = i / 262, col = i - r * 262;
    int yy = y + r - 3, xx = col - 3;
    bfr v = 0;
    if ((unsigned)yy < 256u && (unsigned)xx < 256u) v = plane[(yy << 8) + xx];
    rows[r][col] = v;
  }
  if (tid == 0 && y + 4 < 256) __builtin_prefetch(plane + ((y + 4) << 8), 0, 0);
  __syncthreads();
  float acc = dwb[c];
#pragma unroll
  for (int ky = 0; ky < 7; ++ky)
#pragma unroll
    for (int kx = 0; kx < 7; ++kx)
      acc += sw[ky * 7 + kx] * bf2f(rows[ky][tid + kx]);
  out[((size_t)(b * 128 + c) << 16) + (y << 8) + tid] = f2bf(lrelu(acc));
}

// ---------------------------------------------------------------------------
extern "C" void kernel_launch(void* const* d_in, const int* in_sizes, int n_in,
                              void* d_out, int out_size, void* d_ws, size_t ws_size,
                              hipStream_t stream) {
  const float* x      = (const float*)d_in[0];
  const float* c1w    = (const float*)d_in[1];
  const float* c1b    = (const float*)d_in[2];
  const float* c2w    = (const float*)d_in[3];
  const float* c2b    = (const float*)d_in[4];
  const float* c3w    = (const float*)d_in[5];
  const float* c3b    = (const float*)d_in[6];
  const float* c4w    = (const float*)d_in[7];
  const float* c4b    = (const float*)d_in[8];
  const float* dww1   = (const float*)d_in[9];
  const float* dwg    = (const float*)d_in[10];
  const float* dwbe   = (const float*)d_in[11];
  const float* dww2   = (const float*)d_in[12];
  const float* dwb2   = (const float*)d_in[13];
  const float* dwbias = (const float*)d_in[14];
  const float* c5w    = (const float*)d_in[15];
  const float* c5b    = (const float*)d_in[16];
  (void)in_sizes; (void)n_in; (void)out_size; (void)ws_size;

  char* ws = (char*)d_ws;
  const size_t FEAT_SZ = (size_t)8 * 128 * 256 * 256 * 2;  // bf16 feature map
  bfr* feat  = (bfr*)ws;
  bfr* feat2 = (bfr*)(ws + FEAT_SZ);
  size_t off = 2 * FEAT_SZ;
  bfr* A2 = (bfr*)(ws + off); off += (size_t)32 * 320 * 2;   // 2 mtiles * 10 chunks
  bfr* A3 = (bfr*)(ws + off); off += (size_t)32 * 608 * 2;   // 2 * 19
  bfr* A4 = (bfr*)(ws + off); off += (size_t)32 * 896 * 2;   // 2 * 28
  bfr* A5 = (bfr*)(ws + off); off += (size_t)16 * 1152 * 2;  // 1 * 36
  off = (off + 255) & ~(size_t)255;
  float* pooled = (float*)(ws + off);
  float* wdwbuf = pooled + 1024;

  // up to ~234KB dynamic LDS (CDNA5 WGs may allocate up to 320KB)
  (void)hipFuncSetAttribute((const void*)wmma_conv,
                            hipFuncAttributeMaxDynamicSharedMemorySize,
                            240 * 1024);

  prep_panel<<<(2 * 10 * 512 + 255) / 256, 256, 0, stream>>>(c2w, 32, 33, 2, 10, A2);
  prep_panel<<<(2 * 19 * 512 + 255) / 256, 256, 0, stream>>>(c3w, 32, 65, 2, 19, A3);
  prep_panel<<<(2 * 28 * 512 + 255) / 256, 256, 0, stream>>>(c4w, 31, 97, 2, 28, A4);
  prep_panel<<<(1 * 36 * 512 + 255) / 256, 256, 0, stream>>>(c5w, 3, 128, 1, 36, A5);

  conv1_fused<<<dim3(256, 8), 256, 0, stream>>>(x, c1w, c1b, feat);

  dim3 cg(4, 256, 8);  // 4 x-tiles of 64 px, 256 rows, 8 images
  // shmem = A panel + im2col panel (Kp*128 B) + halo rows
  size_t sh2 = (size_t)32 * 320 * 2 + (size_t)320 * 128 + (size_t)33 * 3 * ROWP * 2;
  size_t sh3 = (size_t)32 * 608 * 2 + (size_t)608 * 128 + (size_t)65 * 3 * ROWP * 2;
  size_t sh4 = (size_t)32 * 896 * 2 + (size_t)896 * 128 + (size_t)97 * 3 * ROWP * 2;
  size_t sh5 = (size_t)16 * 1152 * 2 + (size_t)1152 * 128 + (size_t)128 * 3 * ROWP * 2;

  wmma_conv<<<cg, 256, sh2, stream>>>(feat, 33, 320, A2, c2b, 32, 32, feat, 33, nullptr, 0);
  wmma_conv<<<cg, 256, sh3, stream>>>(feat, 65, 608, A3, c3b, 32, 32, feat, 65, nullptr, 0);
  wmma_conv<<<cg, 256, sh4, stream>>>(feat, 97, 896, A4, c4b, 31, 32, feat, 97, nullptr, 0);

  pool_mean<<<1024, 256, 0, stream>>>(feat, pooled);
  gen_dw<<<1, 256, 0, stream>>>(pooled, dww1, dwg, dwbe, dww2, dwb2, wdwbuf);
  dyn_dw7<<<dim3(256, 128, 8), 256, 0, stream>>>(feat, wdwbuf, dwbias, feat2);

  wmma_conv<<<cg, 256, sh5, stream>>>(feat2, 128, 1152, A5, c5b, 3, 16, nullptr, 0,
                                      (float*)d_out, 3);
}
